// HandWritingSynthesisNet_3298534883972
// MI455X (gfx1250) — matrix operand order repaired and checked by reading, not compiled, not run on hardware
//
#include <hip/hip_runtime.h>
#include <math.h>

// ---------------- problem constants ----------------
#define Bq   64
#define Tq   600
#define Hq   400
#define Vq   77
#define Kq   10
#define Uq   50
#define OUTq 121
#define G4   1600          // 4*H
#define K1P  512           // pad(3+V+H=480)
#define K2P  896           // pad(3+H+V+H=880)
#define KOP  1216          // pad(3H=1200)
#define NOP  128           // pad(OUT=121)
#define NWG  54            // 54 WGs * 8 waves = 432 waves (400 gate tiles + 32 head tiles)
#define NTHR 256

typedef __attribute__((ext_vector_type(16))) __bf16 v16bf;
typedef __attribute__((ext_vector_type(8)))  __bf16 v8bf;
typedef __attribute__((ext_vector_type(8)))  float  v8f;

struct Params {
  const float* inputs; const int* text; const float* tmask;
  const float* w_win;  const float* b_win;
  const __bf16 *W1h, *W1l, *W2h, *W2l, *W3h, *W3l, *WOh, *WOl;
  __bf16 *A1h, *A1l, *A2h, *A2l, *A3h, *A3l, *AOh, *AOl;
  float *g1, *g2, *g3;
  float *h1, *h2, *h3, *c1, *c2, *c3;
  float *mix, *phi, *kap;
  float *bias1, *bias2, *bias3, *bout;
  float *yhat, *wvout, *kapout;
  unsigned *bar;
};

// ---------------- helpers ----------------
__device__ inline __bf16 to_bf16(float f) {
  union { float f; unsigned u; } in; in.f = f;
  unsigned r = (in.u + 0x7FFFu + ((in.u >> 16) & 1u)) >> 16;   // RNE
  union { unsigned short u; __bf16 b; } cv; cv.u = (unsigned short)r;
  return cv.b;
}
__device__ inline float bf2f(__bf16 b) {
  union { __bf16 b; unsigned short u; } s; s.b = b;
  union { unsigned u; float f; } c; c.u = (unsigned)s.u << 16;
  return c.f;
}
// split f into hi + lo bf16 (lo = RNE residual) -> ~fp32-class product accuracy
__device__ inline void split_bf16(float f, __bf16& hi, __bf16& lo) {
  hi = to_bf16(f);
  lo = to_bf16(f - bf2f(hi));
}
__device__ inline float sigf(float x) { return 1.0f / (1.0f + expf(-x)); }

// one 16x16 output tile; split-precision: acc += Ah*Bh + Ah*Bl + Al*Bh
__device__ inline v8f wmma_gemm(const __bf16* __restrict__ Ah,
                                const __bf16* __restrict__ Al, int lda,
                                const __bf16* __restrict__ Bh,
                                const __bf16* __restrict__ Bl, int ldb,
                                int mtile, int ntile, int nkb, int hl, int ln)
{
  v8f acc = {0.f,0.f,0.f,0.f,0.f,0.f,0.f,0.f};
  const size_t aoff = (size_t)(mtile * 16 + ln) * lda;
  const size_t boff = (size_t)(ntile * 16 + ln) * ldb + hl * 16;
  const __bf16* arh = Ah + aoff;
  const __bf16* arl = Al + aoff;
  const __bf16* brh = Bh + boff;
  const __bf16* brl = Bl + boff;
  for (int kb = 0; kb < nkb; ++kb) {
    union { v16bf v; v8bf h[2]; } ah, al, bh, bl;
    int k0 = kb * 32 + hl * 16;
    ah.h[0] = *(const v8bf*)(arh + k0);
    ah.h[1] = *(const v8bf*)(arh + k0 + 8);
    al.h[0] = *(const v8bf*)(arl + k0);
    al.h[1] = *(const v8bf*)(arl + k0 + 8);
    bh.v    = *(const v16bf*)(brh + kb * 32);
    bl.v    = *(const v16bf*)(brl + kb * 32);
    acc = __builtin_amdgcn_wmma_f32_16x16x32_bf16(false, ah.v, false, bh.v,
                                                  (short)0, acc, false, false);
    acc = __builtin_amdgcn_wmma_f32_16x16x32_bf16(false, ah.v, false, bl.v,
                                                  (short)0, acc, false, false);
    acc = __builtin_amdgcn_wmma_f32_16x16x32_bf16(false, al.v, false, bh.v,
                                                  (short)0, acc, false, false);
  }
  return acc;
}

__device__ inline void store_tile_f32(float* out, int ldo, int mtile, int ntile,
                                      int hl, int ln, v8f acc)
{
  int m0 = mtile * 16 + hl * 8;
  int n  = ntile * 16 + ln;
  #pragma unroll
  for (int r = 0; r < 8; ++r) out[(size_t)(m0 + r) * ldo + n] = acc[r];
}

__device__ inline void grid_barrier(unsigned* cnt)
{
  __syncthreads();
  if (threadIdx.x == 0) {
    __threadfence();
    unsigned ticket = atomicAdd(cnt, 1u) + 1u;
    unsigned nwg  = gridDim.x;
    unsigned goal = ((ticket + nwg - 1u) / nwg) * nwg;
    while (__atomic_load_n(cnt, __ATOMIC_RELAXED) < goal) __builtin_amdgcn_s_sleep(1);
    __threadfence();
  }
  __syncthreads();
}

// LSTM pointwise + split-bf16 staging into up to 3 activation panels
__device__ inline void lstm_ewise(int gidx, int stride,
                                  const float* g, const float* bias,
                                  float* c, float* hf,
                                  __bf16* s0h, __bf16* s0l, int ld0, int off0,
                                  __bf16* s1h, __bf16* s1l, int ld1, int off1,
                                  __bf16* s2h, __bf16* s2l, int ld2, int off2)
{
  for (int idx = gidx; idx < Bq * Hq; idx += stride) {
    int b = idx / Hq, j = idx - b * Hq;
    const float* gr = g + (size_t)b * G4;
    float gi = gr[j]            + bias[j];
    float gf = gr[Hq + j]       + bias[Hq + j];
    float gg = gr[2 * Hq + j]   + bias[2 * Hq + j];
    float go = gr[3 * Hq + j]   + bias[3 * Hq + j];
    float cc = sigf(gf) * c[idx] + sigf(gi) * tanhf(gg);
    float hh = sigf(go) * tanhf(cc);
    c[idx] = cc;  hf[idx] = hh;
    __bf16 hb, lb; split_bf16(hh, hb, lb);
    if (s0h) { s0h[(size_t)b * ld0 + off0 + j] = hb; s0l[(size_t)b * ld0 + off0 + j] = lb; }
    if (s1h) { s1h[(size_t)b * ld1 + off1 + j] = hb; s1l[(size_t)b * ld1 + off1 + j] = lb; }
    if (s2h) { s2h[(size_t)b * ld2 + off2 + j] = hb; s2l[(size_t)b * ld2 + off2 + j] = lb; }
  }
}

// ---------------- prep kernels ----------------
__global__ void build_wcat(__bf16* dh, __bf16* dl, const float* wih, int kin,
                           const float* whh, int kh, int kpad, int rows)
{
  int total = rows * kpad;
  for (int idx = blockIdx.x * blockDim.x + threadIdx.x; idx < total;
       idx += gridDim.x * blockDim.x) {
    int g = idx / kpad, k = idx - g * kpad;
    float v = 0.f;
    if (k < kin)            v = wih[(size_t)g * kin + k];
    else if (k < kin + kh)  v = whh[(size_t)g * kh + (k - kin)];
    __bf16 hi, lo; split_bf16(v, hi, lo);
    dh[idx] = hi; dl[idx] = lo;
  }
}

__global__ void build_wout(__bf16* dh, __bf16* dl, const float* w, int nvalid,
                           int kin, int kpad, int rowspad)
{
  int total = rowspad * kpad;
  for (int idx = blockIdx.x * blockDim.x + threadIdx.x; idx < total;
       idx += gridDim.x * blockDim.x) {
    int g = idx / kpad, k = idx - g * kpad;
    float v = (g < nvalid && k < kin) ? w[(size_t)g * kin + k] : 0.f;
    __bf16 hi, lo; split_bf16(v, hi, lo);
    dh[idx] = hi; dl[idx] = lo;
  }
}

__global__ void init_kernel(Params p, const float* h0, const float* c0,
                            const float* pwv, const float* pkap,
                            const float* bih1, const float* bhh1,
                            const float* bih2, const float* bhh2,
                            const float* bih3, const float* bhh3,
                            const float* b_out)
{
  int idx = blockIdx.x * blockDim.x + threadIdx.x;
  const int BH = Bq * Hq;
  if (idx < BH) {
    int b = idx / Hq, j = idx - b * Hq;
    float h1v = h0[idx], h2v = h0[BH + idx], h3v = h0[2 * BH + idx];
    p.h1[idx] = h1v; p.c1[idx] = c0[idx];
    p.h2[idx] = h2v; p.c2[idx] = c0[BH + idx];
    p.h3[idx] = h3v; p.c3[idx] = c0[2 * BH + idx];
    __bf16 h1h, h1l, h2h, h2l, h3h, h3l;
    split_bf16(h1v, h1h, h1l);
    split_bf16(h2v, h2h, h2l);
    split_bf16(h3v, h3h, h3l);
    size_t o1 = (size_t)b * K1P, o2 = (size_t)b * K2P, oo = (size_t)b * KOP;
    p.A1h[o1 + 80  + j] = h1h;  p.A1l[o1 + 80  + j] = h1l;
    p.A2h[o2 + 3   + j] = h1h;  p.A2l[o2 + 3   + j] = h1l;
    p.AOh[oo       + j] = h1h;  p.AOl[oo       + j] = h1l;
    p.A2h[o2 + 480 + j] = h2h;  p.A2l[o2 + 480 + j] = h2l;
    p.A3h[o2 + 3   + j] = h2h;  p.A3l[o2 + 3   + j] = h2l;
    p.AOh[oo + Hq  + j] = h2h;  p.AOl[oo + Hq  + j] = h2l;
    p.A3h[o2 + 480 + j] = h3h;  p.A3l[o2 + 480 + j] = h3l;
    p.AOh[oo + 2*Hq + j] = h3h; p.AOl[oo + 2*Hq + j] = h3l;
  }
  if (idx < Bq * Vq) {
    int b = idx / Vq, v = idx - b * Vq;
    __bf16 wh, wl; split_bf16(pwv[idx], wh, wl);
    size_t o1 = (size_t)b * K1P, o2 = (size_t)b * K2P;
    p.A1h[o1 + 3   + v] = wh;  p.A1l[o1 + 3   + v] = wl;
    p.A2h[o2 + 403 + v] = wh;  p.A2l[o2 + 403 + v] = wl;
    p.A3h[o2 + 403 + v] = wh;  p.A3l[o2 + 403 + v] = wl;
  }
  if (idx < Bq * Kq) p.kap[idx] = pkap[idx];
  if (idx < G4) {
    p.bias1[idx] = bih1[idx] + bhh1[idx];
    p.bias2[idx] = bih2[idx] + bhh2[idx];
    p.bias3[idx] = bih3[idx] + bhh3[idx];
  }
  if (idx < NOP) p.bout[idx] = (idx < OUTq) ? b_out[idx] : 0.f;
  if (idx < Bq * 3) {
    int b = idx / 3, i = idx - b * 3;
    __bf16 xh, xl; split_bf16(p.inputs[((size_t)b * Tq) * 3 + i], xh, xl);  // x_0
    size_t o1 = (size_t)b * K1P, o2 = (size_t)b * K2P;
    p.A1h[o1 + i] = xh;  p.A1l[o1 + i] = xl;
    p.A2h[o2 + i] = xh;  p.A2l[o2 + i] = xl;
    p.A3h[o2 + i] = xh;  p.A3l[o2 + i] = xl;
  }
}

// ---------------- persistent fused kernel ----------------
__global__ void __launch_bounds__(NTHR, 1) hwsynth_main(Params p)
{
  const int tid    = threadIdx.x;
  const int gidx   = blockIdx.x * NTHR + tid;
  const int stride = gridDim.x * NTHR;
  const int wid    = gidx >> 5;
  const int lane   = tid & 31;
  const int hl     = lane >> 4;     // half-wave select
  const int ln     = lane & 15;

  for (int t = 0; t < Tq; ++t) {
    // ---- S1: L1 gate GEMM (A1=[x_t, wv_{t-1}, h1_{t-1}]) + head GEMM for t-1 ----
    if (wid < 400) {
      int mt = wid & 3, nt = wid >> 2;
      v8f acc = wmma_gemm(p.A1h, p.A1l, K1P, p.W1h, p.W1l, K1P,
                          mt, nt, K1P / 32, hl, ln);
      store_tile_f32(p.g1, G4, mt, nt, hl, ln, acc);
    } else if (t > 0) {
      int w2 = wid - 400, mt = w2 & 3, nt = w2 >> 2;
      v8f acc = wmma_gemm(p.AOh, p.AOl, KOP, p.WOh, p.WOl, KOP,
                          mt, nt, KOP / 32, hl, ln);
      int n = nt * 16 + ln;
      if (n < OUTq) {
        float bo = p.bout[n];
        int m0 = mt * 16 + hl * 8;
        #pragma unroll
        for (int r = 0; r < 8; ++r)
          p.yhat[((size_t)(m0 + r) * Tq + (t - 1)) * OUTq + n] = acc[r] + bo;
      }
    }
    grid_barrier(p.bar);

    // ---- S2: L1 pointwise; stage h1 -> A1[80:], A2[3:], AO[0:] ----
    lstm_ewise(gidx, stride, p.g1, p.bias1, p.c1, p.h1,
               p.A1h, p.A1l, K1P, 80,
               p.A2h, p.A2l, K2P, 3,
               p.AOh, p.AOl, KOP, 0);
    grid_barrier(p.bar);

    // ---- S3: attention mix = exp(h1 @ w_win^T + b_win); kappa += dk ----
    if (gidx < Bq * 3 * Kq) {
      int b = gidx / (3 * Kq), j = gidx - b * 3 * Kq;
      const float* hr = p.h1 + (size_t)b * Hq;
      const float* wr = p.w_win + (size_t)j * Hq;
      float s = p.b_win[j];
      for (int k = 0; k < Hq; ++k) s += hr[k] * wr[k];
      float e = expf(s);
      p.mix[b * 3 * Kq + j] = e;
      if (j >= 2 * Kq) p.kap[b * Kq + (j - 2 * Kq)] += e;   // dk
    }
    grid_barrier(p.bar);

    // ---- S4: phi[b,u] ----
    if (gidx < Bq * Uq) {
      int b = gidx / Uq, u = gidx - b * Uq;
      float uu = (float)u, ph = 0.f;
      const float* mb = p.mix + b * 3 * Kq;
      const float* kb = p.kap + b * Kq;
      #pragma unroll
      for (int k = 0; k < Kq; ++k) {
        float d = kb[k] - uu;
        ph += mb[k] * expf(-mb[Kq + k] * d * d);
      }
      p.phi[gidx] = ph * p.tmask[gidx];
    }
    grid_barrier(p.bar);

    // ---- S5: wv[b,v] = sum_u phi * onehot(text); emit wvec; stage wv ----
    if (gidx < Bq * Vq) {
      int b = gidx / Vq, v = gidx - b * Vq;
      const float* pb = p.phi + b * Uq;
      const int*   tb = p.text + b * Uq;
      float s = 0.f;
      for (int u = 0; u < Uq; ++u) if (tb[u] == v) s += pb[u];
      p.wvout[((size_t)b * Tq + t) * Vq + v] = s;
      __bf16 sh, sl; split_bf16(s, sh, sl);
      size_t o1 = (size_t)b * K1P, o2 = (size_t)b * K2P;
      p.A1h[o1 + 3   + v] = sh;  p.A1l[o1 + 3   + v] = sl;
      p.A2h[o2 + 403 + v] = sh;  p.A2l[o2 + 403 + v] = sl;
      p.A3h[o2 + 403 + v] = sh;  p.A3l[o2 + 403 + v] = sl;
    }
    grid_barrier(p.bar);

    // ---- S6: L2 gate GEMM (A2=[x_t, h1_t, wv_t, h2_{t-1}]) ----
    if (wid < 400) {
      int mt = wid & 3, nt = wid >> 2;
      v8f acc = wmma_gemm(p.A2h, p.A2l, K2P, p.W2h, p.W2l, K2P,
                          mt, nt, K2P / 32, hl, ln);
      store_tile_f32(p.g2, G4, mt, nt, hl, ln, acc);
    }
    grid_barrier(p.bar);

    // ---- S7: L2 pointwise; stage h2 -> A2[480:], A3[3:], AO[400:] ----
    lstm_ewise(gidx, stride, p.g2, p.bias2, p.c2, p.h2,
               p.A2h, p.A2l, K2P, 480,
               p.A3h, p.A3l, K2P, 3,
               p.AOh, p.AOl, KOP, Hq);
    grid_barrier(p.bar);

    // ---- S8a: L3 gate GEMM ----
    if (wid < 400) {
      int mt = wid & 3, nt = wid >> 2;
      v8f acc = wmma_gemm(p.A3h, p.A3l, K2P, p.W3h, p.W3l, K2P,
                          mt, nt, K2P / 32, hl, ln);
      store_tile_f32(p.g3, G4, mt, nt, hl, ln, acc);
    }
    grid_barrier(p.bar);

    // ---- S8b: L3 pointwise; stage h3 -> A3[480:], AO[800:]; stage x_{t+1} ----
    lstm_ewise(gidx, stride, p.g3, p.bias3, p.c3, p.h3,
               p.A3h, p.A3l, K2P, 480,
               p.AOh, p.AOl, KOP, 2 * Hq,
               (__bf16*)0, (__bf16*)0, 0, 0);
    if (gidx < Bq * 3 && t + 1 < Tq) {
      int b = gidx / 3, i = gidx - b * 3;
      __bf16 xh, xl;
      split_bf16(p.inputs[((size_t)b * Tq + (t + 1)) * 3 + i], xh, xl);
      size_t o1 = (size_t)b * K1P, o2 = (size_t)b * K2P;
      p.A1h[o1 + i] = xh;  p.A1l[o1 + i] = xl;
      p.A2h[o2 + i] = xh;  p.A2l[o2 + i] = xl;
      p.A3h[o2 + i] = xh;  p.A3l[o2 + i] = xl;
    }
    grid_barrier(p.bar);
  }

  // ---- tail: head GEMM for t = T-1, plus kappa output ----
  if (wid >= 400) {
    int w2 = wid - 400, mt = w2 & 3, nt = w2 >> 2;
    v8f acc = wmma_gemm(p.AOh, p.AOl, KOP, p.WOh, p.WOl, KOP,
                        mt, nt, KOP / 32, hl, ln);
    int n = nt * 16 + ln;
    if (n < OUTq) {
      float bo = p.bout[n];
      int m0 = mt * 16 + hl * 8;
      #pragma unroll
      for (int r = 0; r < 8; ++r)
        p.yhat[((size_t)(m0 + r) * Tq + (Tq - 1)) * OUTq + n] = acc[r] + bo;
    }
  }
  if (gidx < Bq * Kq) p.kapout[gidx] = p.kap[gidx];
}

// ---------------- host launcher ----------------
extern "C" void kernel_launch(void* const* d_in, const int* in_sizes, int n_in,
                              void* d_out, int out_size, void* d_ws, size_t ws_size,
                              hipStream_t stream)
{
  const float* inputs = (const float*)d_in[0];
  const int*   text   = (const int*)  d_in[1];
  const float* tmask  = (const float*)d_in[2];
  const float* h0     = (const float*)d_in[3];
  const float* c0     = (const float*)d_in[4];
  const float* pwv    = (const float*)d_in[5];
  const float* pkap   = (const float*)d_in[6];
  const float* w_ih1  = (const float*)d_in[7];
  const float* w_hh1  = (const float*)d_in[8];
  const float* b_ih1  = (const float*)d_in[9];
  const float* b_hh1  = (const float*)d_in[10];
  const float* w_ih2  = (const float*)d_in[11];
  const float* w_hh2  = (const float*)d_in[12];
  const float* b_ih2  = (const float*)d_in[13];
  const float* b_hh2  = (const float*)d_in[14];
  const float* w_ih3  = (const float*)d_in[15];
  const float* w_hh3  = (const float*)d_in[16];
  const float* b_ih3  = (const float*)d_in[17];
  const float* b_hh3  = (const float*)d_in[18];
  const float* w_win  = (const float*)d_in[19];
  const float* b_win  = (const float*)d_in[20];
  const float* w_out  = (const float*)d_in[21];
  const float* b_out  = (const float*)d_in[22];

  size_t off = 0;
  char* base = (char*)d_ws;
  auto carve = [&](size_t bytes) -> char* {
    char* ptr = base + off;
    off = (off + bytes + 255) & ~(size_t)255;
    return ptr;
  };

  Params P;
  P.W1h = (const __bf16*)carve((size_t)G4 * K1P * 2);
  P.W1l = (const __bf16*)carve((size_t)G4 * K1P * 2);
  P.W2h = (const __bf16*)carve((size_t)G4 * K2P * 2);
  P.W2l = (const __bf16*)carve((size_t)G4 * K2P * 2);
  P.W3h = (const __bf16*)carve((size_t)G4 * K2P * 2);
  P.W3l = (const __bf16*)carve((size_t)G4 * K2P * 2);
  P.WOh = (const __bf16*)carve((size_t)NOP * KOP * 2);
  P.WOl = (const __bf16*)carve((size_t)NOP * KOP * 2);
  P.A1h = (__bf16*)carve((size_t)Bq * K1P * 2);
  P.A1l = (__bf16*)carve((size_t)Bq * K1P * 2);
  P.A2h = (__bf16*)carve((size_t)Bq * K2P * 2);
  P.A2l = (__bf16*)carve((size_t)Bq * K2P * 2);
  P.A3h = (__bf16*)carve((size_t)Bq * K2P * 2);
  P.A3l = (__bf16*)carve((size_t)Bq * K2P * 2);
  P.AOh = (__bf16*)carve((size_t)Bq * KOP * 2);
  P.AOl = (__bf16*)carve((size_t)Bq * KOP * 2);
  P.g1 = (float*)carve((size_t)Bq * G4 * 4);
  P.g2 = (float*)carve((size_t)Bq * G4 * 4);
  P.g3 = (float*)carve((size_t)Bq * G4 * 4);
  P.h1 = (float*)carve((size_t)Bq * Hq * 4);
  P.h2 = (float*)carve((size_t)Bq * Hq * 4);
  P.h3 = (float*)carve((size_t)Bq * Hq * 4);
  P.c1 = (float*)carve((size_t)Bq * Hq * 4);
  P.c2 = (float*)carve((size_t)Bq * Hq * 4);
  P.c3 = (float*)carve((size_t)Bq * Hq * 4);
  P.mix   = (float*)carve((size_t)Bq * 3 * Kq * 4);
  P.phi   = (float*)carve((size_t)Bq * Uq * 4);
  P.kap   = (float*)carve((size_t)Bq * Kq * 4);
  P.bias1 = (float*)carve((size_t)G4 * 4);
  P.bias2 = (float*)carve((size_t)G4 * 4);
  P.bias3 = (float*)carve((size_t)G4 * 4);
  P.bout  = (float*)carve((size_t)NOP * 4);
  P.bar   = (unsigned*)carve(256);
  size_t used = off;

  P.inputs = inputs; P.text = text; P.tmask = tmask;
  P.w_win = w_win;   P.b_win = b_win;
  float* out = (float*)d_out;
  P.yhat   = out;
  P.wvout  = out + (size_t)Bq * Tq * OUTq;
  P.kapout = out + (size_t)Bq * Tq * OUTq + (size_t)Bq * Tq * Vq;

  hipMemsetAsync(d_ws, 0, used, stream);   // zeroes pads, states, barrier counter

  build_wcat<<<256, NTHR, 0, stream>>>((__bf16*)P.W1h, (__bf16*)P.W1l,
                                       w_ih1, 80,  w_hh1, Hq, K1P, G4);
  build_wcat<<<256, NTHR, 0, stream>>>((__bf16*)P.W2h, (__bf16*)P.W2l,
                                       w_ih2, 480, w_hh2, Hq, K2P, G4);
  build_wcat<<<256, NTHR, 0, stream>>>((__bf16*)P.W3h, (__bf16*)P.W3l,
                                       w_ih3, 480, w_hh3, Hq, K2P, G4);
  build_wout<<<256, NTHR, 0, stream>>>((__bf16*)P.WOh, (__bf16*)P.WOl,
                                       w_out, OUTq, 3 * Hq, KOP, NOP);
  init_kernel<<<(Bq * Hq + NTHR - 1) / NTHR, NTHR, 0, stream>>>(
      P, h0, c0, pwv, pkap, b_ih1, b_hh1, b_ih2, b_hh2, b_ih3, b_hh3, b_out);

  hwsynth_main<<<NWG, NTHR, 0, stream>>>(P);
}
